// Gamba_8581344657546
// MI455X (gfx1250) — matrix operand order repaired and checked by reading, not compiled, not run on hardware
//
#include <hip/hip_runtime.h>
#include <hip/hip_bf16.h>

// ---------------- problem constants (match reference) ----------------
#define N_NODES  51200
#define N_EDGES  640000
#define N_GRAPHS 128
#define IN_C     64
#define HID      128
#define OUT_C    64
#define N_TOK    8
#define IDIM     256      // 2*HID
#define SSMN     16
#define DTR      8
#define CONVK    4
#define EPS      1e-5f

typedef __attribute__((ext_vector_type(16))) _Float16 v16h;
typedef __attribute__((ext_vector_type(8)))  _Float16 v8h;
typedef __attribute__((ext_vector_type(8)))  float    v8f;

// ---------------- utility kernels ----------------
__global__ void zero_kernel(float* __restrict__ p, int n) {
    int i = blockIdx.x * blockDim.x + threadIdx.x;
    if (i < n) p[i] = 0.0f;
}

// msg[dst[e]][c] += H[src[e]][c]   (F = feature width, contiguous rows)
// one thread handles 4 consecutive features of one edge
__global__ void scatter_add_kernel(const float* __restrict__ H,
                                   const int* __restrict__ src,
                                   const int* __restrict__ dst,
                                   float* __restrict__ MSG,
                                   int F, int nEdges) {
    int t = blockIdx.x * blockDim.x + threadIdx.x;
    int perEdge = F >> 2;
    int e  = t / perEdge;
    int c4 = (t - e * perEdge) << 2;
    if (e >= nEdges) return;
    int s = src[e], d = dst[e];
    const float4 v = *(const float4*)(H + (size_t)s * F + c4);
    float* o = MSG + (size_t)d * F + c4;
    atomicAdd(o + 0, v.x);
    atomicAdd(o + 1, v.y);
    atomicAdd(o + 2, v.z);
    atomicAdd(o + 3, v.w);
}

// ---------------- WMMA GEMM: OUT[m][n] = sum_k (X[m][k]+MSG[m][k]) * W[n][k] + B[n]
// LDS-staged: block of 256 threads owns a 64-row M-tile.
//  - W (HOUT x HIN) converted to f16 in LDS once per block
//  - A-tile (64 x HIN) = x+msg converted to f16 in LDS
//  - 8 waves: wave -> (n-tile, m-group); B fragment reused across m-subtiles
// grid.x = M/64 ; blockDim.x = 256
template <int HIN, int HOUT>
__global__ void __launch_bounds__(256) gin_gemm_wmma(const float* __restrict__ X,
                                                     const float* __restrict__ MSG,
                                                     const float* __restrict__ W,
                                                     const float* __restrict__ B,
                                                     float* __restrict__ OUT) {
    __shared__ alignas(32) _Float16 Wt[HOUT * HIN];
    __shared__ alignas(32) _Float16 At[64 * HIN];

    const int tid = threadIdx.x;
    const int m0 = blockIdx.x * 64;

    // stage W -> f16 LDS (coalesced float4 loads)
    #pragma unroll 4
    for (int idx = tid * 4; idx < HOUT * HIN; idx += 1024) {
        const float4 w4 = *(const float4*)(W + idx);
        Wt[idx + 0] = (_Float16)w4.x;
        Wt[idx + 1] = (_Float16)w4.y;
        Wt[idx + 2] = (_Float16)w4.z;
        Wt[idx + 3] = (_Float16)w4.w;
    }
    // stage A-tile = X + MSG -> f16 LDS
    #pragma unroll 4
    for (int idx = tid * 4; idx < 64 * HIN; idx += 1024) {
        int r = idx / HIN, c = idx - r * HIN;   // c is 4-aligned (HIN % 4 == 0)
        const float4 xv = *(const float4*)(X + (size_t)(m0 + r) * HIN + c);
        const float4 mv = *(const float4*)(MSG + (size_t)(m0 + r) * HIN + c);
        At[idx + 0] = (_Float16)(xv.x + mv.x);
        At[idx + 1] = (_Float16)(xv.y + mv.y);
        At[idx + 2] = (_Float16)(xv.z + mv.z);
        At[idx + 3] = (_Float16)(xv.w + mv.w);
    }
    __syncthreads();

    const int wave = tid >> 5;
    const int lane = tid & 31;
    const int hs = lane >> 4;      // half-lane select
    const int lm = lane & 15;

    constexpr int nTiles  = HOUT / 16;   // 8 or 4
    constexpr int mGroups = 8 / nTiles;  // 1 or 2
    constexpr int mSubs   = 4 / mGroups; // 4 or 2 (m-subtiles per wave)
    const int ntile = wave % nTiles;
    const int mgrp  = wave / nTiles;
    const int n0 = ntile << 4;

    v8f acc[mSubs];
    #pragma unroll
    for (int s = 0; s < mSubs; ++s) acc[s] = (v8f){};

    #pragma unroll
    for (int k0 = 0; k0 < HIN; k0 += 32) {
        // B fragment (32x16): lane col N=lm ; lanes 0-15 K=k0+j, lanes 16-31 K=k0+16+j
        const v16h b = *(const v16h*)&Wt[(n0 + lm) * HIN + k0 + hs * 16];
        #pragma unroll
        for (int s = 0; s < mSubs; ++s) {
            const int row = (mgrp * mSubs + s) * 16 + lm;
            // A fragment (16x32): elem j K=k0+hs*16+j
            const v16h a = *(const v16h*)&At[row * HIN + k0 + hs * 16];
            acc[s] = __builtin_amdgcn_wmma_f32_16x16x32_f16(
                false, a, false, b, (short)0, acc[s], false, false);
        }
    }

    // C/D layout: lane n=lm, VGPR r -> row M = r + 8*hs
    const int n = n0 + lm;
    const float bias = B[n];
    #pragma unroll
    for (int s = 0; s < mSubs; ++s) {
        #pragma unroll
        for (int r = 0; r < 8; ++r) {
            int m = m0 + (mgrp * mSubs + s) * 16 + r + hs * 8;
            OUT[(size_t)m * HOUT + n] = acc[s][r] + bias;
        }
    }
}

// ---------------- LayerNorm over 128 feats, fused + 0.1*gu (gu identical for all graphs)
__global__ void ln_kernel(const float* __restrict__ gin_out,
                          const float* __restrict__ gu,
                          const float* __restrict__ lw,
                          const float* __restrict__ lb,
                          float* __restrict__ hout) {
    const int node = blockIdx.x;
    const int c = threadIdx.x;  // 128 threads
    __shared__ float red[HID];
    float v = gin_out[(size_t)node * HID + c] + 0.1f * gu[c];
    red[c] = v; __syncthreads();
    for (int s = 64; s > 0; s >>= 1) { if (c < s) red[c] += red[c + s]; __syncthreads(); }
    float mean = red[0] / (float)HID;
    __syncthreads();
    float d = v - mean;
    red[c] = d * d; __syncthreads();
    for (int s = 64; s > 0; s >>= 1) { if (c < s) red[c] += red[c + s]; __syncthreads(); }
    float var = red[0] / (float)HID;
    hout[(size_t)node * HID + c] = d * rsqrtf(var + EPS) * lw[c] + lb[c];
}

// ---------------- final global_add_pool ----------------
__global__ void pool_kernel(const float* __restrict__ hf,
                            const int* __restrict__ batch,
                            float* __restrict__ out) {
    int t = blockIdx.x * blockDim.x + threadIdx.x;  // N_NODES*OUT_C
    if (t >= N_NODES * OUT_C) return;
    int n = t >> 6, c = t & 63;
    atomicAdd(out + batch[n] * OUT_C + c, hf[t]);
}

// ---------------- Mamba over one length-8 sequence (all graphs/tokens identical) ----------------
// Launch: <<<1, 256>>> ; writes gu[128]
__global__ void mamba_kernel(const float* __restrict__ attn_in_b,   // 384
                             const float* __restrict__ attn_out_w,  // 128x128
                             const float* __restrict__ attn_out_b,  // 128
                             const float* __restrict__ norm_w,      // 128
                             const float* __restrict__ in_proj_w,   // 512x128
                             const float* __restrict__ conv_w,      // 256x1x4
                             const float* __restrict__ conv_b,      // 256
                             const float* __restrict__ x_proj_w,    // 40x256
                             const float* __restrict__ dt_proj_w,   // 256x8
                             const float* __restrict__ dt_proj_b,   // 256
                             const float* __restrict__ A_log,       // 256x16
                             const float* __restrict__ Dp,          // 256
                             const float* __restrict__ out_proj_w,  // 128x256
                             const float* __restrict__ norm_f_w,    // 128
                             float* __restrict__ gu) {              // 128 out
    __shared__ float outrow[HID], hn[HID], hout[HID];
    __shared__ float hs0[IDIM], gatev[IDIM];
    __shared__ float hsl[N_TOK][IDIM], dts[N_TOK][IDIM], yv[N_TOK][IDIM];
    __shared__ float Bp[N_TOK][SSMN], Cp[N_TOK][SSMN], tsv[N_TOK][DTR];
    __shared__ float red[256];
    const int tid = threadIdx.x;  // 256 threads

    // out_row = b_v @ attn_out_w^T + attn_out_b  (constant token, every t/graph)
    if (tid < HID) {
        const float* bv = attn_in_b + 2 * HID;
        float acc = attn_out_b[tid];
        for (int k = 0; k < HID; ++k) acc += bv[k] * attn_out_w[tid * HID + k];
        outrow[tid] = acc;
    }
    __syncthreads();

    // RMSNorm
    red[tid] = (tid < HID) ? outrow[tid] * outrow[tid] : 0.0f;
    __syncthreads();
    for (int s = 128; s > 0; s >>= 1) { if (tid < s) red[tid] += red[tid + s]; __syncthreads(); }
    {
        float r = rsqrtf(red[0] / (float)HID + EPS);
        if (tid < HID) hn[tid] = outrow[tid] * norm_w[tid] * r;
    }
    __syncthreads();

    // in_proj: 512 outputs -> hs0[256], gatev[256]
    #pragma unroll
    for (int rep = 0; rep < 2; ++rep) {
        int j = tid + rep * 256;
        float acc = 0.0f;
        for (int k = 0; k < HID; ++k) acc += hn[k] * in_proj_w[j * HID + k];
        if (j < IDIM) hs0[j] = acc; else gatev[j - IDIM] = acc;
    }
    __syncthreads();

    // causal depthwise conv (input constant along t) + SiLU
    {
        const int i = tid;
        float w0 = conv_w[i * CONVK + 0], w1 = conv_w[i * CONVK + 1];
        float w2 = conv_w[i * CONVK + 2], w3 = conv_w[i * CONVK + 3];
        float x = hs0[i], cb = conv_b[i];
        for (int t = 0; t < N_TOK; ++t) {
            float acc = cb + w3 * x;
            if (t >= 1) acc += w2 * x;
            if (t >= 2) acc += w1 * x;
            if (t >= 3) acc += w0 * x;
            hsl[t][i] = acc / (1.0f + __expf(-acc));   // silu
        }
    }
    __syncthreads();

    // x_proj: 8 tokens x 40 outputs
    for (int idx = tid; idx < N_TOK * (DTR + 2 * SSMN); idx += 256) {
        int t = idx / (DTR + 2 * SSMN), j = idx % (DTR + 2 * SSMN);
        float acc = 0.0f;
        for (int k = 0; k < IDIM; ++k) acc += hsl[t][k] * x_proj_w[j * IDIM + k];
        if (j < DTR)            tsv[t][j] = acc;
        else if (j < DTR+SSMN)  Bp[t][j - DTR] = acc;
        else                    Cp[t][j - DTR - SSMN] = acc;
    }
    __syncthreads();

    // dt = softplus(ts @ dt_proj_w^T + b)
    {
        const int i = tid;
        float bb = dt_proj_b[i];
        for (int t = 0; t < N_TOK; ++t) {
            float acc = bb;
            #pragma unroll
            for (int r = 0; r < DTR; ++r) acc += tsv[t][r] * dt_proj_w[i * DTR + r];
            dts[t][i] = (acc > 20.0f) ? acc : __logf(1.0f + __expf(acc));
        }
    }
    __syncthreads();

    // selective scan: thread i owns the 16-wide state for channel i
    {
        const int i = tid;
        float A[SSMN], s[SSMN];
        #pragma unroll
        for (int n = 0; n < SSMN; ++n) { A[n] = -__expf(A_log[i * SSMN + n]); s[n] = 0.0f; }
        float g = gatev[i];
        float gact = g / (1.0f + __expf(-g));
        for (int t = 0; t < N_TOK; ++t) {
            float dt = dts[t][i];
            float du = dt * hsl[t][i];
            float y = 0.0f;
            #pragma unroll
            for (int n = 0; n < SSMN; ++n) {
                s[n] = __expf(dt * A[n]) * s[n] + du * Bp[t][n];
                y += s[n] * Cp[t][n];
            }
            y += hsl[t][i] * Dp[i];
            yv[t][i] = y * gact;
        }
    }
    __syncthreads();

    // out_proj + residual, final RMSNorm, mean over tokens -> gu
    float guacc = 0.0f;
    for (int t = 0; t < N_TOK; ++t) {
        if (tid < HID) {
            float acc = outrow[tid];
            for (int i = 0; i < IDIM; ++i) acc += yv[t][i] * out_proj_w[tid * IDIM + i];
            hout[tid] = acc;
        }
        __syncthreads();
        red[tid] = (tid < HID) ? hout[tid] * hout[tid] : 0.0f;
        __syncthreads();
        for (int s = 128; s > 0; s >>= 1) { if (tid < s) red[tid] += red[tid + s]; __syncthreads(); }
        float r = rsqrtf(red[0] / (float)HID + EPS);
        if (tid < HID) guacc += hout[tid] * norm_f_w[tid] * r;
        __syncthreads();
    }
    if (tid < HID) gu[tid] = guacc / (float)N_TOK;
}

// ---------------- driver ----------------
extern "C" void kernel_launch(void* const* d_in, const int* in_sizes, int n_in,
                              void* d_out, int out_size, void* d_ws, size_t ws_size,
                              hipStream_t stream) {
    (void)in_sizes; (void)n_in; (void)out_size; (void)ws_size;
    // inputs in setup_inputs() insertion order
    const float* x     = (const float*)d_in[0];
    const int*   ei    = (const int*)d_in[1];
    const int*   src   = ei;
    const int*   dst   = ei + N_EDGES;
    const int*   batch = (const int*)d_in[2];
    const float* gin_in_w = (const float*)d_in[3];
    const float* gin_in_b = (const float*)d_in[4];
    // layers: vt, attn_in_b, attn_out_w, attn_out_b, gin_w, gin_b  (x2)
    const float* L_attn_in_b[2]  = { (const float*)d_in[6],  (const float*)d_in[12] };
    const float* L_attn_out_w[2] = { (const float*)d_in[7],  (const float*)d_in[13] };
    const float* L_attn_out_b[2] = { (const float*)d_in[8],  (const float*)d_in[14] };
    const float* L_gin_w[2]      = { (const float*)d_in[9],  (const float*)d_in[15] };
    const float* L_gin_b[2]      = { (const float*)d_in[10], (const float*)d_in[16] };
    // mamba
    const float* m_norm_w    = (const float*)d_in[17];
    const float* m_in_proj_w = (const float*)d_in[18];
    const float* m_conv_w    = (const float*)d_in[19];
    const float* m_conv_b    = (const float*)d_in[20];
    const float* m_x_proj_w  = (const float*)d_in[21];
    const float* m_dt_proj_w = (const float*)d_in[22];
    const float* m_dt_proj_b = (const float*)d_in[23];
    const float* m_A_log     = (const float*)d_in[24];
    const float* m_D         = (const float*)d_in[25];
    const float* m_out_proj_w= (const float*)d_in[26];
    const float* m_norm_f_w  = (const float*)d_in[27];
    const float* gin_out_w   = (const float*)d_in[28];
    const float* gin_out_b   = (const float*)d_in[29];
    const float* ln_w        = (const float*)d_in[30];
    const float* ln_b        = (const float*)d_in[31];

    float* ws  = (float*)d_ws;
    float* msg = ws;                              // N_NODES*128
    float* h   = msg + (size_t)N_NODES * HID;     // N_NODES*128
    float* h2  = h   + (size_t)N_NODES * HID;     // N_NODES*128 (also final N_NODES*64)
    float* gu  = h2  + (size_t)N_NODES * HID;     // 128

    const int ZB = 256;
    auto zgrid = [](int n) { return (n + 255) / 256; };

    // ---- input GIN: h = (x + scatter(x)) @ Win^T + bin   (64 -> 128)
    zero_kernel<<<zgrid(N_NODES * IN_C), ZB, 0, stream>>>(msg, N_NODES * IN_C);
    {
        int thr = N_EDGES * (IN_C / 4);
        scatter_add_kernel<<<(thr + 255) / 256, 256, 0, stream>>>(x, src, dst, msg, IN_C, N_EDGES);
    }
    gin_gemm_wmma<IN_C, HID><<<N_NODES / 64, 256, 0, stream>>>(x, msg, gin_in_w, gin_in_b, h);

    // ---- two Gamba layers
    for (int l = 0; l < 2; ++l) {
        zero_kernel<<<zgrid(N_NODES * HID), ZB, 0, stream>>>(msg, N_NODES * HID);
        {
            int thr = N_EDGES * (HID / 4);
            scatter_add_kernel<<<(thr + 255) / 256, 256, 0, stream>>>(h, src, dst, msg, HID, N_EDGES);
        }
        gin_gemm_wmma<HID, HID><<<N_NODES / 64, 256, 0, stream>>>(h, msg, L_gin_w[l], L_gin_b[l], h2);
        mamba_kernel<<<1, 256, 0, stream>>>(L_attn_in_b[l], L_attn_out_w[l], L_attn_out_b[l],
                                            m_norm_w, m_in_proj_w, m_conv_w, m_conv_b,
                                            m_x_proj_w, m_dt_proj_w, m_dt_proj_b,
                                            m_A_log, m_D, m_out_proj_w, m_norm_f_w, gu);
        ln_kernel<<<N_NODES, HID, 0, stream>>>(h2, gu, ln_w, ln_b, h);
    }

    // ---- output GIN: hf = (h + scatter(h)) @ Wout^T + bout   (128 -> 64)
    zero_kernel<<<zgrid(N_NODES * HID), ZB, 0, stream>>>(msg, N_NODES * HID);
    {
        int thr = N_EDGES * (HID / 4);
        scatter_add_kernel<<<(thr + 255) / 256, 256, 0, stream>>>(h, src, dst, msg, HID, N_EDGES);
    }
    gin_gemm_wmma<HID, OUT_C><<<N_NODES / 64, 256, 0, stream>>>(h, msg, gin_out_w, gin_out_b, h2);

    // ---- global_add_pool
    zero_kernel<<<zgrid(N_GRAPHS * OUT_C), ZB, 0, stream>>>((float*)d_out, N_GRAPHS * OUT_C);
    {
        int thr = N_NODES * OUT_C;
        pool_kernel<<<(thr + 255) / 256, 256, 0, stream>>>(h2, batch, (float*)d_out);
    }
}
